// BahdanauAttention_67920612819170
// MI455X (gfx1250) — matrix operand order repaired and checked by reading, not compiled, not run on hardware
//
#include <hip/hip_runtime.h>
#include <hip/hip_bf16.h>
#include <math.h>

#define B_ 32
#define S_ 2048
#define D_ 1024
#define U_ 1024

#define MT 64                 // rows (M) per block in the fused kernel
#define TPB (S_ / MT)         // 32 M-tiles per batch
#define NKC (D_ / 32)         // 32 k-chunks of 32
#define CPITCH 40             // row pitch within a k-chunk (32 + 8 pad), bf16
#define CSTRIDE (MT * CPITCH + 32)  // 2592 elems: chunk stride + 64B skew

typedef __attribute__((ext_vector_type(16))) __bf16 v16bf;
typedef __attribute__((ext_vector_type(8)))  __bf16 v8bf;
typedef __attribute__((ext_vector_type(4)))  __bf16 v4bf;
typedef __attribute__((ext_vector_type(8)))  float  v8f;

#if __has_builtin(__builtin_amdgcn_tanhf)
#define TANHF(x) __builtin_amdgcn_tanhf(x)     // hw v_tanh_f32 on gfx1250
#elif __has_builtin(__builtin_amdgcn_tanh_f32)
#define TANHF(x) __builtin_amdgcn_tanh_f32(x)
#else
#define TANHF(x) tanhf(x)
#endif

// ---------------------------------------------------------------------------
// Kernel 1: q[b,u] = b2[u] + sum_d query[b,d] * W2[d,u]   (tiny GEMV batch)
// ---------------------------------------------------------------------------
__global__ __launch_bounds__(256)
void q_proj_kernel(const float* __restrict__ query,
                   const float* __restrict__ W2,
                   const float* __restrict__ b2,
                   float* __restrict__ q) {
  const int gid = blockIdx.x * 256 + threadIdx.x;
  const int b = gid / U_;
  const int u = gid % U_;
  float acc = b2[u];
  const float* qr = query + (size_t)b * D_;
#pragma unroll 4
  for (int d = 0; d < D_; ++d)
    acc = fmaf(qr[d], W2[(size_t)d * U_ + u], acc);
  q[gid] = acc;
}

// ---------------------------------------------------------------------------
// Kernel 2: W1 [D,U] fp32 -> W1t [U,D] bf16 (transpose+convert; 2 MB, L2-res)
// ---------------------------------------------------------------------------
__global__ __launch_bounds__(256)
void w1t_kernel(const float* __restrict__ W1, __bf16* __restrict__ w1t) {
  const int idx = blockIdx.x * 256 + threadIdx.x;   // idx = u*D + d
  const int u = idx / D_;
  const int d = idx % D_;
  w1t[idx] = (__bf16)W1[(size_t)d * U_ + u];
}

// ---------------------------------------------------------------------------
// Kernel 3 (main): one 64-row tile of one batch:
//   score_s = tanh(values@W1 + b1 + q) @ V + bv        (bf16 WMMA, K=1024)
//   + local softmax stats + locally-weighted partial context from the
//   bf16 values tile resident in LDS (flash-style; values read ONCE).
// LDS A layout: chunk-major aT[kc][row][40] with 64B chunk skew so all
// fragment addresses are immediate DS offsets (no per-WMMA address VALU,
// no WMMA->VALU WAR hazard NOPs) and reads/stores are bank-conflict-free.
// ---------------------------------------------------------------------------
__global__ __launch_bounds__(256)
void score_kernel(const float* __restrict__ values,
                  const __bf16* __restrict__ w1t,
                  const float* __restrict__ b1,
                  const float* __restrict__ qv,
                  const float* __restrict__ V,
                  const float* __restrict__ bv,
                  float* __restrict__ score,
                  float* __restrict__ stats,        // [B*TPB][2] = m_l, sum_l
                  float* __restrict__ ctxPartial) { // [B*TPB][D]
  extern __shared__ char smem[];
  __bf16* aT   = (__bf16*)smem;                          // NKC * CSTRIDE
  float*  sred = (float*)(smem + NKC * CSTRIDE * 2);     // [8][MT]
  float*  wloc = sred + 8 * MT;                          // [MT]

  const int tid  = threadIdx.x;
  const int lane = tid & 31;
  const int wave = tid >> 5;
  const int b    = blockIdx.x / TPB;
  const int tile = blockIdx.x % TPB;
  const int s0   = tile * MT;
  const int blk  = blockIdx.x;

  // ---- Stage A tile: MT x 1024 fp32 -> bf16 LDS, chunk-major layout
  const float* src = values + ((size_t)b * S_ + s0) * D_;
  for (int i = tid; i < MT * (D_ / 4); i += 256) {
    const int row = i >> 8;            // 256 float4 groups per row
    const int c4  = (i & 255) * 4;
    const int ch  = c4 >> 5;           // k-chunk
    const int kk  = c4 & 31;           // k within chunk
    const float4 v4 = *(const float4*)(src + (size_t)row * D_ + c4);
    v4bf p = {(__bf16)v4.x, (__bf16)v4.y, (__bf16)v4.z, (__bf16)v4.w};
    *(v4bf*)&aT[ch * CSTRIDE + row * CPITCH + kk] = p;
  }
  __syncthreads();

  // A-fragment addressing in contiguous k order (k = 16*half + s), same as B
  const int m  = lane & 15;            // M row within 16-row subtile
  const int kh = (lane >> 4) * 16;     // half-wave K offset
  const int abase = m * CPITCH + kh;   // per-lane base within a chunk

  float part[4][8];
#pragma unroll
  for (int j = 0; j < 4; ++j)
#pragma unroll
    for (int r = 0; r < 8; ++r) part[j][r] = 0.f;

  const float* qb = qv + (size_t)b * U_;

  for (int i = 0; i < U_ / 128; ++i) {          // 8 N-tiles per wave
    const int nt   = i * 128 + wave * 16;
    const int ncol = nt + (lane & 15);
    const __bf16* bp = w1t + (size_t)ncol * D_ + (lane >> 4) * 16;

    v8f acc[4] = {v8f{}, v8f{}, v8f{}, v8f{}};
    for (int kc = 0; kc < NKC; ++kc) {
      // B fragment: contiguous 32B per lane; one B feeds 4 WMMAs.
      v16bf bfrag = *(const v16bf*)(bp + kc * 32);
      const __bf16* ab = &aT[kc * CSTRIDE + abase];
#pragma unroll
      for (int j = 0; j < 4; ++j) {
        // subtile j at immediate offset j*16*CPITCH (<= 1920 elems)
        v8bf a0 = *(const v8bf*)(ab + j * 16 * CPITCH);
        v8bf a1 = *(const v8bf*)(ab + j * 16 * CPITCH + 8);
        v16bf afrag = __builtin_shufflevector(
            a0, a1, 0, 1, 2, 3, 4, 5, 6, 7, 8, 9, 10, 11, 12, 13, 14, 15);
        acc[j] = __builtin_amdgcn_wmma_f32_16x16x32_bf16(
            false, afrag, false, bfrag, (short)0, acc[j], false, false);
      }
    }

    // C layout: VGPR r, lanes 0-15: (M=r, N=lane); lanes 16-31: (M=8+r,...)
    const float add = b1[ncol] + qb[ncol];
    const float vn  = V[ncol];
#pragma unroll
    for (int j = 0; j < 4; ++j)
#pragma unroll
      for (int r = 0; r < 8; ++r)
        part[j][r] += TANHF(acc[j][r] + add) * vn;
  }

  // Reduce over N (16 lanes within each half-wave)
#pragma unroll
  for (int off = 8; off >= 1; off >>= 1)
#pragma unroll
    for (int j = 0; j < 4; ++j)
#pragma unroll
      for (int r = 0; r < 8; ++r)
        part[j][r] += __shfl_xor(part[j][r], off, 32);

  if ((lane & 15) == 0) {
    const int mbase = (lane >> 4) * 8;
#pragma unroll
    for (int j = 0; j < 4; ++j)
#pragma unroll
      for (int r = 0; r < 8; ++r)
        sred[wave * MT + j * 16 + mbase + r] = part[j][r];
  }
  __syncthreads();

  // scores for the block's 64 rows
  if (tid < MT) {
    float s = bv[0];
#pragma unroll
    for (int w = 0; w < 8; ++w) s += sred[w * MT + tid];
    score[(size_t)b * S_ + s0 + tid] = s;
    wloc[tid] = s;
  }
  __syncthreads();

  // local softmax stats (serial over 64 by thread 0: negligible)
  if (tid == 0) {
    float ml = wloc[0];
    for (int s = 1; s < MT; ++s) ml = fmaxf(ml, wloc[s]);
    float sl = 0.f;
    for (int s = 0; s < MT; ++s) {
      float e = __expf(wloc[s] - ml);
      wloc[s] = e;
      sl += e;
    }
    stats[(size_t)blk * 2 + 0] = ml;
    stats[(size_t)blk * 2 + 1] = sl;
  }
  __syncthreads();

  // locally-weighted partial context from the LDS values tile
  const int d0  = tid * 4;
  const int ch0 = d0 >> 5;
  const int kk0 = d0 & 31;
  float c0 = 0.f, c1 = 0.f, c2 = 0.f, c3 = 0.f;
  for (int s = 0; s < MT; ++s) {
    const float w = wloc[s];
    v4bf v = *(const v4bf*)&aT[ch0 * CSTRIDE + s * CPITCH + kk0];
    c0 = fmaf(w, (float)v[0], c0);
    c1 = fmaf(w, (float)v[1], c1);
    c2 = fmaf(w, (float)v[2], c2);
    c3 = fmaf(w, (float)v[3], c3);
  }
  float4 outv = {c0, c1, c2, c3};
  *(float4*)&ctxPartial[(size_t)blk * D_ + d0] = outv;
}

// ---------------------------------------------------------------------------
// Kernel 4: per-batch combine of 32 tile-partials (flash-style rescale)
// ---------------------------------------------------------------------------
__global__ __launch_bounds__(256)
void combine_kernel(const float* __restrict__ stats,
                    const float* __restrict__ ctxPartial,
                    const float* __restrict__ score,
                    float* __restrict__ ctx_out,
                    float* __restrict__ attn_out) {
  __shared__ float scale[TPB];
  __shared__ float gStats[2];  // M, 1/Sum
  const int b = blockIdx.x;
  const int tid = threadIdx.x;

  if (tid == 0) {
    float M = -3.402823466e38f;
    for (int t = 0; t < TPB; ++t)
      M = fmaxf(M, stats[((size_t)b * TPB + t) * 2]);
    float Sum = 0.f;
    for (int t = 0; t < TPB; ++t) {
      const float e = __expf(stats[((size_t)b * TPB + t) * 2] - M);
      scale[t] = e;
      Sum += e * stats[((size_t)b * TPB + t) * 2 + 1];
    }
    gStats[0] = M;
    gStats[1] = 1.f / Sum;
  }
  __syncthreads();

  const float M = gStats[0];
  const float inv = gStats[1];

  for (int d = tid; d < D_; d += 256) {
    float a = 0.f;
#pragma unroll 4
    for (int t = 0; t < TPB; ++t)
      a = fmaf(scale[t], ctxPartial[((size_t)b * TPB + t) * D_ + d], a);
    ctx_out[(size_t)b * D_ + d] = a * inv;
  }

  for (int s = tid; s < S_; s += 256)
    attn_out[(size_t)b * S_ + s] = __expf(score[(size_t)b * S_ + s] - M) * inv;
}

// ---------------------------------------------------------------------------
extern "C" void kernel_launch(void* const* d_in, const int* in_sizes, int n_in,
                              void* d_out, int out_size, void* d_ws, size_t ws_size,
                              hipStream_t stream) {
  const float* query  = (const float*)d_in[0];
  const float* values = (const float*)d_in[1];
  const float* W1     = (const float*)d_in[2];
  const float* b1     = (const float*)d_in[3];
  const float* W2     = (const float*)d_in[4];
  const float* b2     = (const float*)d_in[5];
  const float* V      = (const float*)d_in[6];
  const float* bv     = (const float*)d_in[7];

  float* out      = (float*)d_out;
  float* ctx_out  = out;              // [B, D]
  float* attn_out = out + B_ * D_;    // [B, S, 1]

  // Workspace layout (w1t offset stays 32B-aligned for v16bf loads):
  float*  q          = (float*)d_ws;            // B*U floats
  float*  score      = q + B_ * U_;             // B*S floats
  float*  stats      = score + B_ * S_;         // B*TPB*2 floats
  float*  ctxPartial = stats + B_ * TPB * 2;    // B*TPB*D floats (4 MB)
  __bf16* w1t        = (__bf16*)(ctxPartial + (size_t)B_ * TPB * D_);

  const size_t shmem = (size_t)NKC * CSTRIDE * 2    // bf16 A tile (162 KB)
                     + (size_t)(8 * MT + MT) * 4;   // sred + wloc

  q_proj_kernel<<<(B_ * U_) / 256, 256, 0, stream>>>(query, W2, b2, q);
  w1t_kernel<<<(D_ * U_) / 256, 256, 0, stream>>>(W1, w1t);
  score_kernel<<<B_ * TPB, 256, shmem, stream>>>(values, w1t, b1, q, V, bv,
                                                 score, stats, ctxPartial);
  combine_kernel<<<B_, 256, 0, stream>>>(stats, ctxPartial, score, ctx_out,
                                         attn_out);
}
